// CACombiner_40793599377808
// MI455X (gfx1250) — matrix operand order repaired and checked by reading, not compiled, not run on hardware
//
#include <hip/hip_runtime.h>
#include <hip/hip_bf16.h>

// ---------------------------------------------------------------------------
// CDNA5 (gfx1250) fused "efficient attention + FFN" block.
// All GEMMs run on v_wmma_f32_16x16x32_bf16 with f32 accumulate.
// Tile staging uses GLOBAL_LOAD_ASYNC_TO_LDS_B128 + double-buffered LDS when
// the toolchain exposes the async builtins (guarded by __has_builtin).
// ---------------------------------------------------------------------------

typedef __bf16 bf16_t;
typedef __attribute__((ext_vector_type(16))) __bf16 v16bf;
typedef __attribute__((ext_vector_type(8)))  float  v8f;
typedef __attribute__((ext_vector_type(4)))  float  f32x4;
typedef __attribute__((vector_size(16)))     int    v4i_t;

#if __has_builtin(__builtin_amdgcn_global_load_async_to_lds_b128)
#define HAVE_ASYNC_LDS 1
#else
#define HAVE_ASYNC_LDS 0
#endif

static constexpr int Bn  = 8;
static constexpr int Cn  = 512;
static constexpr int Ln  = 4096;
static constexpr int KCn = 512;
static constexpr int VCn = 512;
static constexpr int Hn  = 8;
static constexpr int C2n = 1024;      // 2*C
static constexpr long BLn = (long)Bn * Ln;

union Frag {
    v16bf v;
    f32x4 f4[2];
};

__device__ __forceinline__ float elu_act(float v) {
    return v > 0.f ? v : (expf(v) - 1.f);
}

__device__ __forceinline__ void wait_async0() {
#if HAVE_ASYNC_LDS
#if __has_builtin(__builtin_amdgcn_s_wait_asynccnt)
    __builtin_amdgcn_s_wait_asynccnt(0);
#else
    asm volatile("s_wait_asynccnt 0x0" ::: "memory");
#endif
#endif
}

__device__ __forceinline__ void async_cp16(bf16_t* lds, const bf16_t* g) {
#if HAVE_ASYNC_LDS
    __builtin_amdgcn_global_load_async_to_lds_b128(
        (__attribute__((address_space(1))) v4i_t*)(uintptr_t)g,
        (__attribute__((address_space(3))) v4i_t*)lds,
        0, 0);
#endif
}

// ---------------------------------------------------------------------------
// Generic bf16 WMMA GEMM:  D[m,n] = sum_k A[m,k] * Bt[n,k]   (+epilogue)
// A, Bt row-major bf16.  Batched via blockIdx.z = b1*8 + b2 with separate
// element strides per (b1, b2) so (batch, head) splits work uniformly.
// Double-buffered LDS; async global->LDS DMA when available.
// ---------------------------------------------------------------------------
template <int BM, int BN, int BK, int WR, int WC,
          bool BIASM, bool BIASN, bool RESID, bool ELU, bool OUT32>
__global__ __launch_bounds__(WR * WC * 32)
void gemm_bf16_wmma(const bf16_t* __restrict__ A,  int ldA, long sA1, long sA2,
                    const bf16_t* __restrict__ Bt, int ldB, long sB1, long sB2,
                    void* __restrict__ Dp,         int ldD, long sD1, long sD2,
                    const float* __restrict__ biasM,
                    const float* __restrict__ biasN,
                    const float* __restrict__ resid, int resLd, long sR2,
                    int M, int N, int K)
{
    constexpr int THREADS = WR * WC * 32;
    constexpr int SUBM = BM / (WR * 16);
    constexpr int SUBN = BN / (WC * 16);
    constexpr int LDS_STRIDE = BK + 8;          // halves; pad to spread banks
    constexpr int CPR = BK / 8;                 // 16B chunks per tile row
    constexpr int ITA = (BM * CPR) / THREADS;   // exact (all shapes divide)
    constexpr int ITB = (BN * CPR) / THREADS;

    __shared__ alignas(16) bf16_t lA[2][BM * LDS_STRIDE];
    __shared__ alignas(16) bf16_t lB[2][BN * LDS_STRIDE];

    const int bz = blockIdx.z;
    const int b1 = bz >> 3;
    const int b2 = bz & 7;
    A  += b1 * sA1 + b2 * sA2;
    Bt += b1 * sB1 + b2 * sB2;

    const int m0 = blockIdx.y * BM;
    const int n0 = blockIdx.x * BN;

    const int tid  = threadIdx.x;
    const int lane = tid & 31;
    const int wid  = tid >> 5;
    const int wm   = wid / WC;
    const int wn   = wid % WC;
    const int l15  = lane & 15;
    const int sel  = lane >> 4;

    v8f acc[SUBM][SUBN];
    const v8f vzero = {0.f, 0.f, 0.f, 0.f, 0.f, 0.f, 0.f, 0.f};
#pragma unroll
    for (int i = 0; i < SUBM; ++i)
#pragma unroll
        for (int j = 0; j < SUBN; ++j) acc[i][j] = vzero;

    // Stage one K-tile (A: BM x BK, Bt: BN x BK) into LDS buffer `buf`.
    auto stage = [&](int buf, int k0) {
#if HAVE_ASYNC_LDS
#pragma unroll
        for (int it = 0; it < ITA; ++it) {
            const int ci = tid + it * THREADS;
            const int r = ci / CPR, c = (ci % CPR) * 8;
            async_cp16(&lA[buf][r * LDS_STRIDE + c],
                       &A[(long)(m0 + r) * ldA + k0 + c]);
        }
#pragma unroll
        for (int it = 0; it < ITB; ++it) {
            const int ci = tid + it * THREADS;
            const int r = ci / CPR, c = (ci % CPR) * 8;
            async_cp16(&lB[buf][r * LDS_STRIDE + c],
                       &Bt[(long)(n0 + r) * ldB + k0 + c]);
        }
#else
        f32x4 ta[ITA], tb[ITB];
#pragma unroll
        for (int it = 0; it < ITA; ++it) {
            const int ci = tid + it * THREADS;
            const int r = ci / CPR, c = (ci % CPR) * 8;
            ta[it] = *(const f32x4*)&A[(long)(m0 + r) * ldA + k0 + c];
        }
#pragma unroll
        for (int it = 0; it < ITB; ++it) {
            const int ci = tid + it * THREADS;
            const int r = ci / CPR, c = (ci % CPR) * 8;
            tb[it] = *(const f32x4*)&Bt[(long)(n0 + r) * ldB + k0 + c];
        }
#pragma unroll
        for (int it = 0; it < ITA; ++it) {
            const int ci = tid + it * THREADS;
            const int r = ci / CPR, c = (ci % CPR) * 8;
            *(f32x4*)&lA[buf][r * LDS_STRIDE + c] = ta[it];
        }
#pragma unroll
        for (int it = 0; it < ITB; ++it) {
            const int ci = tid + it * THREADS;
            const int r = ci / CPR, c = (ci % CPR) * 8;
            *(f32x4*)&lB[buf][r * LDS_STRIDE + c] = tb[it];
        }
#endif
    };
    auto commit = [&]() {
        wait_async0();       // own async DMA done (no-op in sync path)
        __syncthreads();     // all waves' tile writes visible
    };

    stage(0, 0);
    commit();

    int buf = 0;
    for (int k0 = 0; k0 < K; k0 += BK) {
        const int kn = k0 + BK;
        if (kn < K) stage(buf ^ 1, kn);       // overlap DMA with compute

#pragma unroll
        for (int kk = 0; kk < BK; kk += 32) {
            Frag af[SUBM];
            Frag bfr[SUBN];
            // A fragment: lane half sel -> K {sel*8..+7} and {sel*8+16..+23}
#pragma unroll
            for (int i = 0; i < SUBM; ++i) {
                const int row = wm * (SUBM * 16) + i * 16 + l15;
                const bf16_t* p = &lA[buf][row * LDS_STRIDE + kk + sel * 8];
                af[i].f4[0] = *(const f32x4*)p;
                af[i].f4[1] = *(const f32x4*)(p + 16);
            }
            // B fragment: lane half sel -> K {sel*8..+7} and {sel*8+16..+23}
#pragma unroll
            for (int j = 0; j < SUBN; ++j) {
                const int row = wn * (SUBN * 16) + j * 16 + l15;
                const bf16_t* p = &lB[buf][row * LDS_STRIDE + kk + sel * 8];
                bfr[j].f4[0] = *(const f32x4*)p;
                bfr[j].f4[1] = *(const f32x4*)(p + 16);
            }
#pragma unroll
            for (int i = 0; i < SUBM; ++i)
#pragma unroll
                for (int j = 0; j < SUBN; ++j)
                    acc[i][j] = __builtin_amdgcn_wmma_f32_16x16x32_bf16(
                        false, af[i].v, false, bfr[j].v,
                        (short)0, acc[i][j], false, false);
        }

        if (kn < K) {
            commit();
            buf ^= 1;
        }
    }

    // Epilogue: D tile layout — VGPR r: lanes0-15 M=r, lanes16-31 M=r+8.
    const long dOff = b1 * sD1 + b2 * sD2;
    float* Df = (float*)Dp + dOff;
    bf16_t* Db = (bf16_t*)Dp + dOff;
    if (RESID) resid += b2 * sR2;

#pragma unroll
    for (int i = 0; i < SUBM; ++i) {
#pragma unroll
        for (int j = 0; j < SUBN; ++j) {
            const int col = n0 + wn * (SUBN * 16) + j * 16 + l15;
            float bn = 0.f;
            if (BIASN) bn = biasN[col];
#pragma unroll
            for (int r = 0; r < 8; ++r) {
                const int row = m0 + wm * (SUBM * 16) + i * 16 + sel * 8 + r;
                float v = acc[i][j][r];
                if (BIASM) v += biasM[row];
                if (BIASN) v += bn;
                if (RESID) v += resid[(long)col * resLd + row];
                if (ELU)   v = elu_act(v);
                if (OUT32) Df[(long)row * ldD + col] = v;
                else       Db[(long)row * ldD + col] = (bf16_t)v;
            }
        }
    }
}

// ---------------------------------------------------------------------------
// Elementwise f32 -> bf16 weight conversion
// ---------------------------------------------------------------------------
__global__ __launch_bounds__(256)
void cvt_f32_bf16(const float* __restrict__ in, bf16_t* __restrict__ out, int n)
{
    int i = blockIdx.x * 256 + threadIdx.x;
    if (i < n) out[i] = (bf16_t)in[i];
}

// ---------------------------------------------------------------------------
// Tiled transpose: src [P,Q] f32 (batched) -> dst [Q,P] (bf16 or f32)
// ---------------------------------------------------------------------------
template <bool OUTBF>
__global__ __launch_bounds__(256)
void transpose_kernel(const float* __restrict__ src, void* __restrict__ dstp,
                      int P, int Q)
{
    __shared__ float t[32][33];
    const long batch = (long)blockIdx.z * P * Q;
    src += batch;
    const int q0 = blockIdx.x * 32;
    const int p0 = blockIdx.y * 32;
    const int tx = threadIdx.x, ty = threadIdx.y;
#pragma unroll
    for (int k = 0; k < 32; k += 8)
        t[ty + k][tx] = src[(long)(p0 + ty + k) * Q + q0 + tx];
    __syncthreads();
#pragma unroll
    for (int k = 0; k < 32; k += 8) {
        const float v = t[tx][ty + k];
        const long o = batch + (long)(q0 + ty + k) * P + p0 + tx;
        if (OUTBF) ((bf16_t*)dstp)[o] = (bf16_t)v;
        else       ((float*)dstp)[o]  = v;
    }
}

// ---------------------------------------------------------------------------
// Row softmax over length L (keys): bf16 in/out, one block per row.
// ---------------------------------------------------------------------------
__global__ __launch_bounds__(256)
void softmax_rows(const bf16_t* __restrict__ in, bf16_t* __restrict__ out, int n)
{
    __shared__ float red[256];
    const long row = blockIdx.x;
    in  += row * n;
    out += row * n;
    const int tid = threadIdx.x;

    float m = -3.4e38f;
    for (int i = tid; i < n; i += 256) m = fmaxf(m, (float)in[i]);
    red[tid] = m; __syncthreads();
    for (int o = 128; o > 0; o >>= 1) {
        if (tid < o) red[tid] = fmaxf(red[tid], red[tid + o]);
        __syncthreads();
    }
    m = red[0]; __syncthreads();

    float s = 0.f;
    for (int i = tid; i < n; i += 256) s += expf((float)in[i] - m);
    red[tid] = s; __syncthreads();
    for (int o = 128; o > 0; o >>= 1) {
        if (tid < o) red[tid] += red[tid + o];
        __syncthreads();
    }
    const float inv = 1.f / red[0];

    for (int i = tid; i < n; i += 256)
        out[i] = (bf16_t)(expf((float)in[i] - m) * inv);
}

// ---------------------------------------------------------------------------
// Query softmax over DK=64 (channel dim within head), writes transposed:
// qht[b, l, h*64+dk]. One thread per (b,h,l).
// ---------------------------------------------------------------------------
__global__ __launch_bounds__(256)
void softmax_q_kernel(const bf16_t* __restrict__ q, bf16_t* __restrict__ qht)
{
    const long idx = (long)blockIdx.x * 256 + threadIdx.x;   // over B*H*L
    const long l  = idx % Ln;
    const long bh = idx / Ln;
    const bf16_t* src = q + (bh * 64) * (long)Ln + l;        // stride L per dk

    float vals[64];
    float m = -3.4e38f;
#pragma unroll
    for (int i = 0; i < 64; ++i) {
        vals[i] = (float)src[(long)i * Ln];
        m = fmaxf(m, vals[i]);
    }
    float s = 0.f;
#pragma unroll
    for (int i = 0; i < 64; ++i) { vals[i] = expf(vals[i] - m); s += vals[i]; }
    const float inv = 1.f / s;

    const long b = bh >> 3, h = bh & 7;
    bf16_t* dst = qht + (b * (long)Ln + l) * KCn + h * 64;
#pragma unroll
    for (int i = 0; i < 64; ++i) dst[i] = (bf16_t)(vals[i] * inv);
}

// ---------------------------------------------------------------------------
// LayerNorm over C=512 per token row; out bf16 or f32. One block per token.
// ---------------------------------------------------------------------------
template <bool OUTBF>
__global__ __launch_bounds__(256)
void ln_kernel(const float* __restrict__ x, void* __restrict__ outp,
               const float* __restrict__ g, const float* __restrict__ be)
{
    __shared__ float red[256];
    const long row = blockIdx.x;
    x += row * Cn;
    const int tid = threadIdx.x;

    const float a0 = x[tid];
    const float a1 = x[tid + 256];

    red[tid] = a0 + a1; __syncthreads();
    for (int o = 128; o > 0; o >>= 1) {
        if (tid < o) red[tid] += red[tid + o];
        __syncthreads();
    }
    const float mean = red[0] * (1.f / Cn);
    __syncthreads();

    red[tid] = a0 * a0 + a1 * a1; __syncthreads();
    for (int o = 128; o > 0; o >>= 1) {
        if (tid < o) red[tid] += red[tid + o];
        __syncthreads();
    }
    const float var = red[0] * (1.f / Cn) - mean * mean;
    const float rstd = rsqrtf(var + 1e-5f);

    const float y0 = (a0 - mean) * rstd * g[tid]       + be[tid];
    const float y1 = (a1 - mean) * rstd * g[tid + 256] + be[tid + 256];
    if (OUTBF) {
        ((bf16_t*)outp)[row * Cn + tid]       = (bf16_t)y0;
        ((bf16_t*)outp)[row * Cn + tid + 256] = (bf16_t)y1;
    } else {
        ((float*)outp)[row * Cn + tid]       = y0;
        ((float*)outp)[row * Cn + tid + 256] = y1;
    }
}

// ---------------------------------------------------------------------------
extern "C" void kernel_launch(void* const* d_in, const int* in_sizes, int n_in,
                              void* d_out, int out_size, void* d_ws, size_t ws_size,
                              hipStream_t stream)
{
    const float* z1  = (const float*)d_in[0];
    const float* z2  = (const float*)d_in[1];
    const float* Wq  = (const float*)d_in[2];
    const float* bq  = (const float*)d_in[3];
    const float* Wk  = (const float*)d_in[4];
    const float* bk  = (const float*)d_in[5];
    const float* Wv  = (const float*)d_in[6];
    const float* bv  = (const float*)d_in[7];
    const float* Wr  = (const float*)d_in[8];
    const float* br  = (const float*)d_in[9];
    const float* g1  = (const float*)d_in[10];
    const float* be1 = (const float*)d_in[11];
    const float* W1  = (const float*)d_in[12];
    const float* b1  = (const float*)d_in[13];
    const float* W2  = (const float*)d_in[14];
    const float* b2  = (const float*)d_in[15];
    const float* g2  = (const float*)d_in[16];
    const float* be2 = (const float*)d_in[17];

    // ---- workspace bump allocator (deterministic) ----
    char* wp = (char*)d_ws;
    auto alloc = [&](size_t bytes) -> char* {
        char* p = wp;
        wp += (bytes + 255) & ~(size_t)255;
        return p;
    };
    const size_t SZ_W    = (size_t)KCn * Cn * 2;          // 512 KB
    const size_t SZ_ZT   = (size_t)Bn * Ln * Cn * 2;      // 32 MB bf16
    const size_t SZ_F32  = (size_t)Bn * Ln * Cn * 4;      // 64 MB f32

    bf16_t* Wq_b = (bf16_t*)alloc(SZ_W);
    bf16_t* Wk_b = (bf16_t*)alloc(SZ_W);
    bf16_t* Wv_b = (bf16_t*)alloc(SZ_W);
    bf16_t* Wr_b = (bf16_t*)alloc(SZ_W);
    bf16_t* W1_b = (bf16_t*)alloc((size_t)C2n * Cn * 2);
    bf16_t* W2_b = (bf16_t*)alloc((size_t)Cn * C2n * 2);

    bf16_t* z1T = (bf16_t*)alloc(SZ_ZT);
    bf16_t* z2T = (bf16_t*)alloc(SZ_ZT);
    bf16_t* qb  = (bf16_t*)alloc(SZ_ZT);      // raw q [B,KC,L]
    bf16_t* kb  = (bf16_t*)alloc(SZ_ZT);      // raw k [B,KC,L] (contig after qb)
    bf16_t* vb  = (bf16_t*)alloc(SZ_ZT);      // raw v [B,VC,L]
    bf16_t* khs = (bf16_t*)alloc(SZ_ZT);      // softmaxed k
    bf16_t* qht = (bf16_t*)alloc(SZ_ZT);      // softmaxed q, [B,L,KC]
    bf16_t* ctxT = (bf16_t*)alloc((size_t)Bn * Hn * 64 * 64 * 2);
    float*  zT  = (float*)alloc(SZ_F32);      // [B,L,C] pre-LN1
    bf16_t* xb  = (bf16_t*)alloc(SZ_ZT);      // post-LN1 bf16

    // Aliases (regions dead by the time they are reused):
    bf16_t* attT = z1T;                       // [B,L,VC]  (z1T dead after q GEMM)
    bf16_t* hb   = qb;                        // [BL,2C] bf16 -> uses qb+kb (64MB)
    float*  yf   = zT;                        // FFN2 output reuses zT
    float*  lnT  = (float*)qb;                // LN2 output [B,L,C] f32 (64MB)

    const dim3 blk256(256), blkT(32, 8);

    // 1) weight conversions
    cvt_f32_bf16<<<dim3((KCn*Cn + 255)/256), blk256, 0, stream>>>(Wq, Wq_b, KCn*Cn);
    cvt_f32_bf16<<<dim3((KCn*Cn + 255)/256), blk256, 0, stream>>>(Wk, Wk_b, KCn*Cn);
    cvt_f32_bf16<<<dim3((KCn*Cn + 255)/256), blk256, 0, stream>>>(Wv, Wv_b, KCn*Cn);
    cvt_f32_bf16<<<dim3((Cn*VCn + 255)/256), blk256, 0, stream>>>(Wr, Wr_b, Cn*VCn);
    cvt_f32_bf16<<<dim3((C2n*Cn + 255)/256), blk256, 0, stream>>>(W1, W1_b, C2n*Cn);
    cvt_f32_bf16<<<dim3((Cn*C2n + 255)/256), blk256, 0, stream>>>(W2, W2_b, Cn*C2n);

    // 2) transpose-convert activations: z[B,C,L] f32 -> zT[B,L,C] bf16
    transpose_kernel<true><<<dim3(Ln/32, Cn/32, Bn), blkT, 0, stream>>>(z1, z1T, Cn, Ln);
    transpose_kernel<true><<<dim3(Ln/32, Cn/32, Bn), blkT, 0, stream>>>(z2, z2T, Cn, Ln);

    // 3) q/k/v projections: D[o,l] = W[o,:] . zT[l,:]  (bias over rows)
    {
        const long sB = (long)Ln * Cn, sD = (long)KCn * Ln;
        gemm_bf16_wmma<128,128,64,4,2, true,false,false,false,false>
            <<<dim3(Ln/128, KCn/128, Bn), blk256, 0, stream>>>(
            Wq_b, Cn, 0, 0,  z1T, Cn, 8*sB, sB,
            qb, Ln, 8*sD, sD, bq, nullptr, nullptr, 0, 0, KCn, Ln, Cn);
        gemm_bf16_wmma<128,128,64,4,2, true,false,false,false,false>
            <<<dim3(Ln/128, KCn/128, Bn), blk256, 0, stream>>>(
            Wk_b, Cn, 0, 0,  z2T, Cn, 8*sB, sB,
            kb, Ln, 8*sD, sD, bk, nullptr, nullptr, 0, 0, KCn, Ln, Cn);
        gemm_bf16_wmma<128,128,64,4,2, true,false,false,false,false>
            <<<dim3(Ln/128, VCn/128, Bn), blk256, 0, stream>>>(
            Wv_b, Cn, 0, 0,  z2T, Cn, 8*sB, sB,
            vb, Ln, 8*sD, sD, bv, nullptr, nullptr, 0, 0, VCn, Ln, Cn);
    }

    // 4) softmaxes
    softmax_rows<<<dim3(Bn * KCn), blk256, 0, stream>>>(kb, khs, Ln);
    softmax_q_kernel<<<dim3((Bn * Hn * Ln) / 256), blk256, 0, stream>>>(qb, qht);

    // 5) ctxT[b,h][v,k] = sum_l v[v,l] * ksm[k,l]   (per-head 64x64, K=L)
    {
        const long sA = (long)64 * Ln;      // per (b,h), uniform
        const long sD = 64 * 64;
        gemm_bf16_wmma<64,64,64,2,2, false,false,false,false,false>
            <<<dim3(1, 1, Bn * Hn), dim3(128), 0, stream>>>(
            vb, Ln, 8*sA, sA,  khs, Ln, 8*sA, sA,
            ctxT, 64, 8*sD, sD, nullptr, nullptr, nullptr, 0, 0, 64, 64, Ln);
    }

    // 6) attT[b][l, h*64+v] = sum_k qht[b][l, h*64+k] * ctxT[b,h][v,k]
    gemm_bf16_wmma<128,64,64,4,2, false,false,false,false,false>
        <<<dim3(1, Ln/128, Bn * Hn), blk256, 0, stream>>>(
        qht, KCn, (long)Ln * KCn, 64,
        ctxT, 64, 8L * 64 * 64, 64 * 64,
        attT, VCn, (long)Ln * VCn, 64,
        nullptr, nullptr, nullptr, 0, 0, Ln, 64, 64);

    // 7) reprojection + residual: zT[b][l,c] = attT[l,:].Wr[c,:] + br[c] + z1[b,c,l]
    {
        const long sA = (long)Ln * VCn, sD = (long)Ln * Cn;
        gemm_bf16_wmma<128,128,64,4,2, false,true,true,false,true>
            <<<dim3(Cn/128, Ln/128, Bn), blk256, 0, stream>>>(
            attT, VCn, 8*sA, sA,  Wr_b, VCn, 0, 0,
            zT, Cn, 8*sD, sD, nullptr, br, z1, Ln, (long)Cn * Ln, Ln, Cn, VCn);
    }

    // 8) LN1 -> xb bf16
    ln_kernel<true><<<dim3((unsigned)BLn), blk256, 0, stream>>>(zT, xb, g1, be1);

    // 9) FFN1: hb = elu(xb . W1^T + b1)   [BL, 2C] bf16
    gemm_bf16_wmma<128,128,64,4,2, false,true,false,true,false>
        <<<dim3(C2n/128, (unsigned)(BLn/128), 1), blk256, 0, stream>>>(
        xb, Cn, 0, 0,  W1_b, Cn, 0, 0,
        hb, C2n, 0, 0, nullptr, b1, nullptr, 0, 0, (int)BLn, C2n, Cn);

    // 10) FFN2: yf = hb . W2^T + b2       [BL, C] f32 (reuses zT)
    gemm_bf16_wmma<128,128,64,4,2, false,true,false,false,true>
        <<<dim3(Cn/128, (unsigned)(BLn/128), 1), blk256, 0, stream>>>(
        hb, C2n, 0, 0,  W2_b, C2n, 0, 0,
        yf, Cn, 0, 0, nullptr, b2, nullptr, 0, 0, (int)BLn, Cn, C2n);

    // 11) LN2 -> lnT f32 [B,L,C]
    ln_kernel<false><<<dim3((unsigned)BLn), blk256, 0, stream>>>(yf, lnT, g2, be2);

    // 12) transpose back: lnT [B,L,C] -> out [B,C,L] f32
    transpose_kernel<false><<<dim3(Cn/32, Ln/32, Bn), blkT, 0, stream>>>(
        lnT, d_out, Ln, Cn);
}
